// FlashAttention_73349451481221
// MI455X (gfx1250) — matrix operand order repaired and checked by reading, not compiled, not run on hardware
//
#include <hip/hip_runtime.h>
#include <hip/hip_bf16.h>

typedef __attribute__((ext_vector_type(16))) _Float16 v16h;
typedef __attribute__((ext_vector_type(8)))  _Float16 v8h;
typedef __attribute__((ext_vector_type(8)))  float    v8f;
typedef __attribute__((__vector_size__(4 * sizeof(int)))) int v4i;

#define BATCH 2
#define SEQ   2048
#define DIM   1024
#define NH    16
#define HD2   64
#define NTOK  (BATCH * SEQ)   // 4096

#define GLOBAL_AS __attribute__((address_space(1)))
#define LDS_AS    __attribute__((address_space(3)))

#if defined(__has_builtin)
#if __has_builtin(__builtin_amdgcn_global_load_async_to_lds_b128)
#define HAVE_ASYNC_LDS 1
#endif
#endif

static __device__ inline v8f wmma_f16(v16h a, v16h b, v8f c) {
  return __builtin_amdgcn_wmma_f32_16x16x32_f16(false, a, false, b, (short)0, c,
                                                false, false);
}

// A-fragment (16x32 f16): lane<16 holds row M=lane, halves K=koff..koff+7 and
// K=koff+16..koff+23 with koff = (lane>=16 ? 8 : 0).  Two 16B loads.
static __device__ inline v16h load_frag_split(const _Float16* p) {
  v8h lo = *(const v8h*)p;
  v8h hi = *(const v8h*)(p + 16);
  v16h r;
#pragma unroll
  for (int i = 0; i < 8; ++i) { r[i] = lo[i]; r[8 + i] = hi[i]; }
  return r;
}

static __device__ inline v8f zero8() {
  v8f z;
#pragma unroll
  for (int i = 0; i < 8; ++i) z[i] = 0.f;
  return z;
}

// copy 16 bytes global -> LDS (async DMA if available, VGPR bounce otherwise)
static __device__ inline void copy16_g2l(const _Float16* g, _Float16* l) {
#ifdef HAVE_ASYNC_LDS
  __builtin_amdgcn_global_load_async_to_lds_b128(
      (GLOBAL_AS v4i*)g, (LDS_AS v4i*)l, 0, 0);
#else
  *(v8h*)l = *(const v8h*)g;
#endif
}

static __device__ inline void wait_g2l() {
#ifdef HAVE_ASYNC_LDS
  asm volatile("s_wait_asynccnt 0x0" ::: "memory");
#endif
}

// ---------------- elementwise f32 -> f16 convert ----------------
__global__ void cvt_f32_to_f16(const float* __restrict__ in,
                               _Float16* __restrict__ out, int n) {
  int i = blockIdx.x * blockDim.x + threadIdx.x;
  if (i < n) out[i] = (_Float16)in[i];
}

// ---------------- tiled transpose + convert: Wt[n][k] = W[k][n] ----------------
__global__ void transpose_cvt(const float* __restrict__ w,
                              _Float16* __restrict__ wt) {
  __shared__ float tile[32][33];
  int tx = threadIdx.x, ty = threadIdx.y;           // (32, 8)
  int bx = blockIdx.x * 32, by = blockIdx.y * 32;
#pragma unroll
  for (int t = 0; t < 4; ++t)
    tile[ty + 8 * t][tx] = w[(by + ty + 8 * t) * DIM + bx + tx];
  __syncthreads();
#pragma unroll
  for (int t = 0; t < 4; ++t)
    wt[(bx + ty + 8 * t) * DIM + by + tx] = (_Float16)tile[tx][ty + 8 * t];
}

// ---------------- WMMA GEMM: C[m,n] = sum_k A[m,k]*Bt[n,k] + bias[n] ----------
// Block = 8 waves (256 thr) sharing one 64-row A tile (double-buffered in LDS
// via async copies).  Each wave computes a 64(M) x 32(N) tile -> 8 WMMAs per
// k-step, 8 f32 accumulators (no spills).  Block covers 64 x 256.
// mode: 0=Q store, 1=K store, 2=V store (transposed [b,h,hd,s]), 3=f32 out.
__global__ void __launch_bounds__(256)
gemm_wmma(const _Float16* __restrict__ A,
          const _Float16* __restrict__ Bt,
          const float* __restrict__ bias,
          _Float16* __restrict__ outh,
          float* __restrict__ outf,
          int M, int K, int N, int mode) {
  __shared__ _Float16 Atile[2][64 * 32];            // 2 x 4 KB

  int tid  = threadIdx.x;                           // 0..255
  int lane = tid & 31;
  int wave = tid >> 5;                              // 0..7
  int gw   = blockIdx.x * 8 + wave;
  int wavesN = N / 32;                              // 8 consecutive gw share m0
  int m0 = (gw / wavesN) * 64;
  int n0 = (gw % wavesN) * 32;

  int hi  = lane >> 4;
  int l16 = lane & 15;
  int akoff = hi << 3;                              // 0 or 8 halves
  int bkoff = hi << 3;                              // 0 or 8 halves

  // cooperative copy of one 64x32 A chunk (4 KB = 256 x 16B, 256 threads)
  int crow = tid >> 2;                              // 0..63
  int ccol = (tid & 3) * 8;                         // halves
  auto copyA = [&](int buf, int kk) {
    copy16_g2l(A + (size_t)(m0 + crow) * K + kk + ccol,
               &Atile[buf][crow * 32 + ccol]);
  };

  v8f acc[4][2];
#pragma unroll
  for (int i = 0; i < 4; ++i)
#pragma unroll
    for (int j = 0; j < 2; ++j) acc[i][j] = zero8();

  copyA(0, 0);
  wait_g2l();
  __syncthreads();

  int cur = 0;
  for (int kk = 0; kk < K; kk += 32) {
    if (kk + 32 < K) copyA(cur ^ 1, kk + 32);       // prefetch next chunk

    v16h bf[2];
#pragma unroll
    for (int j = 0; j < 2; ++j) {
      const _Float16* pb = Bt + (size_t)(n0 + 16 * j + l16) * K + kk + bkoff;
      bf[j] = load_frag_split(pb);
    }
    v16h af[4];
#pragma unroll
    for (int i = 0; i < 4; ++i)
      af[i] = load_frag_split(&Atile[cur][(16 * i + l16) * 32 + akoff]);

#pragma unroll
    for (int i = 0; i < 4; ++i)
#pragma unroll
      for (int j = 0; j < 2; ++j) acc[i][j] = wmma_f16(af[i], bf[j], acc[i][j]);

    wait_g2l();
    __syncthreads();
    cur ^= 1;
  }

  float bia[2];
#pragma unroll
  for (int j = 0; j < 2; ++j) bia[j] = bias[n0 + 16 * j + l16];

  // C/D layout: VGPR r -> M = r + 8*hi (within 16-tile), N = lane&15.
#pragma unroll
  for (int i = 0; i < 4; ++i)
#pragma unroll
    for (int j = 0; j < 2; ++j)
#pragma unroll
      for (int r = 0; r < 8; ++r) {
        int m = m0 + 16 * i + r + (hi << 3);
        int n = n0 + 16 * j + l16;
        float v = acc[i][j][r] + bia[j];
        if (mode == 3) {
          outf[(size_t)m * N + n] = v;
        } else {
          int b = m >> 11, s = m & (SEQ - 1);
          int h = n >> 6,  hd = n & (HD2 - 1);
          if (mode == 2)   // V transposed: [b,h,hd,s]
            outh[(((size_t)(b * NH + h) * HD2 + hd) * SEQ) + s] = (_Float16)v;
          else             // Q/K: [b,h,s,hd]
            outh[(((size_t)(b * NH + h) * SEQ + s) * HD2) + hd] = (_Float16)v;
        }
      }
}

// ---------------- flash attention core ----------------
// One wave per (b, h, 16-query tile).  Key tiles of 32.
__global__ void __launch_bounds__(128)
flash_attn(const _Float16* __restrict__ Q,
           const _Float16* __restrict__ Kt,  // [b,h,s,hd]
           const _Float16* __restrict__ Vt,  // [b,h,hd,s]
           _Float16* __restrict__ attn) {    // [b*s, D]
  __shared__ _Float16 Plds[4][16 * 32];

  int tid  = threadIdx.x;
  int lane = tid & 31;
  int wave = tid >> 5;
  int gw   = blockIdx.x * 4 + wave;
  int qt   = gw & (SEQ / 16 - 1);
  int bh   = gw / (SEQ / 16);
  int h    = bh & (NH - 1);
  int b    = bh / NH;
  int qs   = qt * 16;

  int hi  = lane >> 4;
  int l16 = lane & 15;

  const _Float16* Qb = Q + (size_t)(bh * SEQ) * HD2;
  const _Float16* Kb = Kt + (size_t)(bh * SEQ) * HD2;
  const _Float16* Vb = Vt + (size_t)(bh * HD2) * SEQ;

  const _Float16* qp = Qb + (size_t)(qs + l16) * HD2 + (hi << 3);
  v16h a0 = load_frag_split(qp);
  v16h a1 = load_frag_split(qp + 32);

  float mi[8], li[8];
  v8f oacc[4];
#pragma unroll
  for (int r = 0; r < 8; ++r) { mi[r] = -3.0e38f; li[r] = 0.f; }
#pragma unroll
  for (int j = 0; j < 4; ++j) oacc[j] = zero8();

  const float scale = 0.125f;   // 1/sqrt(64)
  int ktMax = (qs + 15) >> 5;

  for (int kt = 0; kt <= ktMax; ++kt) {
    int ks = kt << 5;
    v8f sc[2];
#pragma unroll
    for (int n = 0; n < 2; ++n) {
      const _Float16* kp = Kb + (size_t)(ks + 16 * n + l16) * HD2 + (hi << 3);
      v16h b0 = load_frag_split(kp);
      v16h b1 = load_frag_split(kp + 32);
      v8f c = zero8();
      c = wmma_f16(a0, b0, c);
      c = wmma_f16(a1, b1, c);
      sc[n] = c;
    }

#pragma unroll
    for (int r = 0; r < 8; ++r) {
      int mg = qs + r + (hi << 3);
      int n0g = ks + l16;
      float s0 = sc[0][r] * scale;
      float s1 = sc[1][r] * scale;
      if (n0g > mg)      s0 = -3.0e38f;
      if (n0g + 16 > mg) s1 = -3.0e38f;
      float tmax = fmaxf(s0, s1);
#pragma unroll
      for (int off = 1; off < 16; off <<= 1)
        tmax = fmaxf(tmax, __shfl_xor(tmax, off, 32));
      float mnew = fmaxf(mi[r], tmax);
      float corr = __expf(mi[r] - mnew);
      float p0 = __expf(s0 - mnew);
      float p1 = __expf(s1 - mnew);
      float rs = p0 + p1;
#pragma unroll
      for (int off = 1; off < 16; off <<= 1)
        rs += __shfl_xor(rs, off, 32);
      li[r] = li[r] * corr + rs;
      mi[r] = mnew;
#pragma unroll
      for (int j = 0; j < 4; ++j) oacc[j][r] *= corr;
      int prow = r + (hi << 3);
      Plds[wave][prow * 32 + l16]      = (_Float16)p0;
      Plds[wave][prow * 32 + l16 + 16] = (_Float16)p1;
    }
    asm volatile("s_wait_dscnt 0x0" ::: "memory");

    const _Float16* pp = &Plds[wave][l16 * 32 + (hi << 3)];
    v16h pA = load_frag_split(pp);

#pragma unroll
    for (int j = 0; j < 4; ++j) {
      const _Float16* vp = Vb + (size_t)(16 * j + l16) * SEQ + ks + (hi << 3);
      v16h bv = load_frag_split(vp);
      oacc[j] = wmma_f16(pA, bv, oacc[j]);
    }
  }

#pragma unroll
  for (int j = 0; j < 4; ++j)
#pragma unroll
    for (int r = 0; r < 8; ++r) {
      int m = r + (hi << 3);
      float o = oacc[j][r] / li[r];
      size_t row = (size_t)(b * SEQ + qs + m);
      attn[row * DIM + h * HD2 + 16 * j + l16] = (_Float16)o;
    }
}

extern "C" void kernel_launch(void* const* d_in, const int* in_sizes, int n_in,
                              void* d_out, int out_size, void* d_ws, size_t ws_size,
                              hipStream_t stream) {
  const float* x  = (const float*)d_in[0];
  const float* wq = (const float*)d_in[1];
  const float* bq = (const float*)d_in[2];
  const float* wk = (const float*)d_in[3];
  const float* bk = (const float*)d_in[4];
  const float* wv = (const float*)d_in[5];
  const float* bv = (const float*)d_in[6];
  const float* wo = (const float*)d_in[7];
  const float* bo = (const float*)d_in[8];
  float* out = (float*)d_out;

  char* ws = (char*)d_ws;
  size_t off = 0;
  _Float16* xh  = (_Float16*)(ws + off); off += (size_t)NTOK * DIM * 2;
  _Float16* wqt = (_Float16*)(ws + off); off += (size_t)DIM * DIM * 2;
  _Float16* wkt = (_Float16*)(ws + off); off += (size_t)DIM * DIM * 2;
  _Float16* wvt = (_Float16*)(ws + off); off += (size_t)DIM * DIM * 2;
  _Float16* wot = (_Float16*)(ws + off); off += (size_t)DIM * DIM * 2;
  _Float16* Qh  = (_Float16*)(ws + off); off += (size_t)NTOK * DIM * 2;
  _Float16* Kh  = (_Float16*)(ws + off); off += (size_t)NTOK * DIM * 2;
  _Float16* Vh  = (_Float16*)(ws + off); off += (size_t)NTOK * DIM * 2;
  _Float16* attn = (_Float16*)(ws + off); off += (size_t)NTOK * DIM * 2;

  {
    int n = NTOK * DIM;
    cvt_f32_to_f16<<<(n + 255) / 256, 256, 0, stream>>>(x, xh, n);
  }
  {
    dim3 blk(32, 8), grd(DIM / 32, DIM / 32);
    transpose_cvt<<<grd, blk, 0, stream>>>(wq, wqt);
    transpose_cvt<<<grd, blk, 0, stream>>>(wk, wkt);
    transpose_cvt<<<grd, blk, 0, stream>>>(wv, wvt);
    transpose_cvt<<<grd, blk, 0, stream>>>(wo, wot);
  }
  {
    int waves = (NTOK / 64) * (DIM / 32);           // 2048 waves
    int blocks = waves / 8;                         // 256 blocks
    gemm_wmma<<<blocks, 256, 0, stream>>>(xh, wqt, bq, Qh, nullptr,
                                          NTOK, DIM, DIM, 0);
    gemm_wmma<<<blocks, 256, 0, stream>>>(xh, wkt, bk, Kh, nullptr,
                                          NTOK, DIM, DIM, 1);
    gemm_wmma<<<blocks, 256, 0, stream>>>(xh, wvt, bv, Vh, nullptr,
                                          NTOK, DIM, DIM, 2);
  }
  {
    int waves = BATCH * NH * (SEQ / 16);            // 4096 waves
    flash_attn<<<waves / 4, 128, 0, stream>>>(Qh, Kh, Vh, attn);
  }
  {
    int waves = (NTOK / 64) * (DIM / 32);
    gemm_wmma<<<waves / 8, 256, 0, stream>>>(attn, wot, bo, nullptr, out,
                                             NTOK, DIM, DIM, 3);
  }
}
